// GeometryConstrainedAttention_1322849927237
// MI455X (gfx1250) — compile-verified, matrix operand order checked by reading, not run on hardware
//
#include <hip/hip_runtime.h>
#include <hip/hip_bf16.h>

// ---------------------------------------------------------------------------
// GeometryConstrainedAttention for MI455X (gfx1250, wave32, WMMA bf16)
//   E=512, H=8, D=64, B=4, Nq=Nk=2048
// - All matmuls on v_wmma_f32_16x16x32_bf16.
// - Tiles staged in LDS pre-swizzled into WMMA fragment layout; Q/K/V staging
//   uses CDNA5 GLOBAL_LOAD_ASYNC_TO_LDS_B128 (ASYNCcnt-tracked), with K/V
//   DOUBLE-BUFFERED so the DMA for tile kt+1 overlaps compute on tile kt.
// - V is produced pre-transposed [B,H,D,N] by its projection so the PV
//   B-operand stages with contiguous async copies (no LDS scatter).
// - Mask is transposed once into [B,Nk,Nq] (L2-resident) so the masked-softmax
//   reads one aligned 8B global load per tile per lane.
// ---------------------------------------------------------------------------

typedef __attribute__((ext_vector_type(16))) __bf16 v16bf;
typedef __attribute__((ext_vector_type(8)))  __bf16 v8bf;
typedef __attribute__((ext_vector_type(8)))  float  v8f;
typedef __attribute__((ext_vector_type(4)))  int    v4i;

#define E_DIM 512
#define H_HEADS 8
#define D_HEAD 64
#define B_BATCH 4
#define N_SEQ 2048

#if __has_builtin(__builtin_amdgcn_global_load_async_to_lds_b128)
#define HAVE_ASYNC_LDS 1
typedef __attribute__((address_space(1))) v4i* gv4i_ptr;
typedef __attribute__((address_space(3))) v4i* lv4i_ptr;
#else
#define HAVE_ASYNC_LDS 0
#endif

__device__ __forceinline__ void async_copy16(const __bf16* gsrc, __bf16* ldst) {
#if HAVE_ASYNC_LDS
  __builtin_amdgcn_global_load_async_to_lds_b128(
      (gv4i_ptr)(__bf16*)gsrc, (lv4i_ptr)ldst, 0, 0);
#else
  *(v8bf*)ldst = *(const v8bf*)gsrc;
#endif
}

__device__ __forceinline__ void async_wait() {
#if HAVE_ASYNC_LDS
#if __has_builtin(__builtin_amdgcn_s_wait_asynccnt)
  __builtin_amdgcn_s_wait_asynccnt(0);
#else
  asm volatile("s_wait_asynccnt 0x0" ::: "memory");
#endif
#endif
}

__device__ __forceinline__ v8f wmma_bf16(v16bf a, v16bf b, v8f c) {
  return __builtin_amdgcn_wmma_f32_16x16x32_bf16(
      /*neg_a=*/false, a, /*neg_b=*/false, b,
      /*c_mod=*/(short)0, c, /*reuse_a=*/false, /*reuse_b=*/false);
}

// Flat offset of element (rn, k) inside a fragment region [32 lanes][16 bf16].
// lane = rn + 16*k[3], elem = k[2:0] + 8*k[4].
// For k0 aligned to 8, (rn, k0..k0+7) is CONTIGUOUS at frag_off(rn, k0).
__device__ __forceinline__ int frag_off(int rn, int k) {
  return (rn + (((k >> 3) & 1) << 4)) * 16 + (k & 7) + (((k >> 4) & 1) << 3);
}

// ---------------------------------------------------------------------------
// GEMM + bias: C[8192,512] = A[8192,512] @ W[512,512] + bias
// OUT_MODE: 0 -> f32 row-major, 1 -> bf16 [B,H,N,D], 2 -> bf16 [B,H,D,N]
// Block = 128 threads (4 waves), tile 64x64, K step 32.
// ---------------------------------------------------------------------------
template <bool A_BF16, int OUT_MODE>
__global__ __launch_bounds__(128) void gemm_bias_kernel(
    const void* __restrict__ Aptr, const float* __restrict__ W,
    const float* __restrict__ bias, void* __restrict__ outp) {
  __shared__ v16bf AsF[4][32];  // [m-block][lane]
  __shared__ v16bf WsF[4][32];  // [n-block][lane]

  const int m0 = blockIdx.y * 64;
  const int n0 = blockIdx.x * 64;
  const int lane = threadIdx.x & 31;
  const int wave = threadIdx.x >> 5;

  v8f acc[4] = {v8f{}, v8f{}, v8f{}, v8f{}};

  for (int ks = 0; ks < E_DIM; ks += 32) {
    __syncthreads();
    // Stage A: 64 rows x 4 groups-of-8 k (2 tasks/thread, fully unrolled)
#pragma unroll
    for (int it = 0; it < 2; ++it) {
      const int t = threadIdx.x + it * 128;
      const int r = t >> 2, k0 = (t & 3) * 8;
      __bf16* dst = (__bf16*)&AsF[r >> 4][0] + frag_off(r & 15, k0);
      if (A_BF16) {
        async_copy16((const __bf16*)Aptr + (size_t)(m0 + r) * E_DIM + ks + k0, dst);
      } else {
        const float* ap = (const float*)Aptr + (size_t)(m0 + r) * E_DIM + ks + k0;
        const float4 f0 = *(const float4*)ap;
        const float4 f1 = *(const float4*)(ap + 4);
        v8bf val;
        val[0] = (__bf16)f0.x; val[1] = (__bf16)f0.y;
        val[2] = (__bf16)f0.z; val[3] = (__bf16)f0.w;
        val[4] = (__bf16)f1.x; val[5] = (__bf16)f1.y;
        val[6] = (__bf16)f1.z; val[7] = (__bf16)f1.w;
        *(v8bf*)dst = val;
      }
    }
    // Stage W: 32 k-rows x 8 groups-of-8 n (f32->bf16, 8 b16 scatter)
#pragma unroll
    for (int it = 0; it < 2; ++it) {
      const int t = threadIdx.x + it * 128;
      const int k = t >> 3, nn0 = (t & 7) * 8;
      const float* wp = W + (size_t)(ks + k) * E_DIM + n0 + nn0;
      const float4 f0 = *(const float4*)wp;
      const float4 f1 = *(const float4*)(wp + 4);
      const float fv[8] = {f0.x, f0.y, f0.z, f0.w, f1.x, f1.y, f1.z, f1.w};
      __bf16* dst = (__bf16*)&WsF[nn0 >> 4][0];
#pragma unroll
      for (int j = 0; j < 8; ++j)
        dst[frag_off((nn0 + j) & 15, k)] = (__bf16)fv[j];
    }
    if (A_BF16) async_wait();
    __syncthreads();

    const v16bf a = AsF[wave][lane];
#pragma unroll
    for (int t = 0; t < 4; ++t)
      acc[t] = wmma_bf16(a, WsF[t][lane], acc[t]);
  }

  const int hi = lane >> 4, n = lane & 15;
#pragma unroll
  for (int t = 0; t < 4; ++t) {
#pragma unroll
    for (int i = 0; i < 8; ++i) {
      const int row = m0 + wave * 16 + i + 8 * hi;
      const int col = n0 + t * 16 + n;
      const float v = acc[t][i] + bias[col];
      const int bb = row >> 11, nq = row & (N_SEQ - 1);
      const int hh = col >> 6, dd = col & 63;
      if (OUT_MODE == 1) {
        ((__bf16*)outp)[(((size_t)(bb * H_HEADS + hh)) * N_SEQ + nq) * D_HEAD + dd] =
            (__bf16)v;
      } else if (OUT_MODE == 2) {
        ((__bf16*)outp)[(((size_t)(bb * H_HEADS + hh)) * D_HEAD + dd) * N_SEQ + nq] =
            (__bf16)v;
      } else {
        ((float*)outp)[(size_t)row * E_DIM + col] = v;
      }
    }
  }
}

// ---------------------------------------------------------------------------
// One-shot mask transpose: mask[B][Nq][Nk] (u8) -> maskT[B][Nk][Nq]
// Tiled through LDS, coalesced 4B on both sides. grid (Nk/64, Nq/64, B).
// ---------------------------------------------------------------------------
__global__ __launch_bounds__(128) void mask_transpose_kernel(
    const unsigned char* __restrict__ m, unsigned char* __restrict__ mt) {
  __shared__ unsigned char tile[64][68];
  const int kx = blockIdx.x * 64, qy = blockIdx.y * 64, b = blockIdx.z;
  const unsigned char* src = m + ((size_t)b * N_SEQ + qy) * N_SEQ + kx;
#pragma unroll
  for (int it = 0; it < 8; ++it) {
    const int t = threadIdx.x + it * 128;
    const int r = t >> 4, c0 = (t & 15) * 4;  // r = q-local, c0 = k-local
    const unsigned int w = *(const unsigned int*)(src + (size_t)r * N_SEQ + c0);
    tile[c0 + 0][r] = (unsigned char)(w & 0xff);
    tile[c0 + 1][r] = (unsigned char)((w >> 8) & 0xff);
    tile[c0 + 2][r] = (unsigned char)((w >> 16) & 0xff);
    tile[c0 + 3][r] = (unsigned char)((w >> 24) & 0xff);
  }
  __syncthreads();
  unsigned char* dst = mt + ((size_t)b * N_SEQ + kx) * N_SEQ + qy;
#pragma unroll
  for (int it = 0; it < 8; ++it) {
    const int t = threadIdx.x + it * 128;
    const int r = t >> 4, c0 = (t & 15) * 4;  // r = k-local, c0 = q-local
    const unsigned int w = (unsigned int)tile[r][c0] |
                           ((unsigned int)tile[r][c0 + 1] << 8) |
                           ((unsigned int)tile[r][c0 + 2] << 16) |
                           ((unsigned int)tile[r][c0 + 3] << 24);
    *(unsigned int*)(dst + (size_t)r * N_SEQ + c0) = w;
  }
}

// ---------------------------------------------------------------------------
// Stage one 64x64 K tile and one 64x64 V^T tile (async) into fragment-layout
// LDS buffers.  Both sides of every copy are contiguous 16B.
// ---------------------------------------------------------------------------
__device__ __forceinline__ void stage_kv_tiles(const __bf16* __restrict__ Kg,
                                               const __bf16* __restrict__ VgT,
                                               int kt, int tid,
                                               v16bf (*KsB)[2][32],
                                               v16bf (*VsB)[2][32]) {
#pragma unroll
  for (int it = 0; it < 4; ++it) {
    const int t = tid + it * 128;
    const int j = t >> 3, d0 = (t & 7) * 8;
    async_copy16(Kg + ((size_t)kt * 64 + j) * 64 + d0,
                 (__bf16*)&KsB[j >> 4][d0 >> 5][0] + frag_off(j & 15, d0 & 31));
  }
#pragma unroll
  for (int it = 0; it < 4; ++it) {
    const int t = tid + it * 128;
    const int d = t >> 3, j0 = (t & 7) * 8;
    async_copy16(VgT + (size_t)d * N_SEQ + (size_t)kt * 64 + j0,
                 (__bf16*)&VsB[d >> 4][j0 >> 5][0] + frag_off(d & 15, j0 & 31));
  }
}

// ---------------------------------------------------------------------------
// Flash attention. grid = (Nq/64, H, B), block = 128 (4 waves).
// VbT is pre-transposed [B,H,D,N]; maskT is [B,Nk,Nq].
// K/V double-buffered: async DMA for tile kt+1 overlaps compute on tile kt.
// ---------------------------------------------------------------------------
__global__ __launch_bounds__(128) void attn_kernel(
    const __bf16* __restrict__ Qb, const __bf16* __restrict__ Kb,
    const __bf16* __restrict__ VbT, const unsigned char* __restrict__ maskT,
    __bf16* __restrict__ Ob) {
  const int qt = blockIdx.x;
  const int h  = blockIdx.y;
  const int b  = blockIdx.z;
  const int lane = threadIdx.x & 31;
  const int wave = threadIdx.x >> 5;
  const int hi = lane >> 4, ln = lane & 15;
  const float scale = 0.125f;  // D^-0.5

  __shared__ v16bf QsF[4][2][32];     // A-op: Q rows,    k = d
  __shared__ v16bf KsF[2][4][2][32];  // B-op: n = K row, k = d   (double buf)
  __shared__ v16bf VsF[2][4][2][32];  // B-op: n = d,     k = K row (double buf)
  __shared__ v16bf PsF[4][2][32];     // A-op: P rows,    k = K row

  const size_t bh = (size_t)(b * H_HEADS + h);
  const __bf16* Qg = Qb + (bh * N_SEQ + (size_t)qt * 64) * D_HEAD;
  const __bf16* Kg = Kb + bh * N_SEQ * D_HEAD;
  const __bf16* VgT = VbT + bh * D_HEAD * N_SEQ;  // rows d, stride N_SEQ
  const unsigned char* MgT = maskT + (size_t)b * N_SEQ * N_SEQ + (size_t)qt * 64;

  // Stage Q (async): 64 rows x 8 d-groups, 4 tasks/thread
#pragma unroll
  for (int it = 0; it < 4; ++it) {
    const int t = threadIdx.x + it * 128;
    const int r = t >> 3, d0 = (t & 7) * 8;
    async_copy16(Qg + (size_t)r * 64 + d0,
                 (__bf16*)&QsF[r >> 4][d0 >> 5][0] + frag_off(r & 15, d0 & 31));
  }
  // Prologue: also kick off K/V for tile 0 into buffer 0
  stage_kv_tiles(Kg, VgT, 0, threadIdx.x, KsF[0], VsF[0]);
  async_wait();
  __syncthreads();

  const v16bf aQ0 = QsF[wave][0][lane];
  const v16bf aQ1 = QsF[wave][1][lane];

  float mrow[8], lrow[8];
#pragma unroll
  for (int i = 0; i < 8; ++i) { mrow[i] = -1e30f; lrow[i] = 0.0f; }
  v8f accO[4] = {v8f{}, v8f{}, v8f{}, v8f{}};

  for (int kt = 0; kt < N_SEQ / 64; ++kt) {
    const int cur = kt & 1;
    if (kt > 0) {
      async_wait();      // my async copies into buf[cur] complete
      __syncthreads();   // everyone's complete; buf[cur^1] no longer being read
    }
    // Overlap: DMA the next K/V tiles while we compute on the current ones
    if (kt + 1 < N_SEQ / 64)
      stage_kv_tiles(Kg, VgT, kt + 1, threadIdx.x, KsF[cur ^ 1], VsF[cur ^ 1]);

    // S = Q K^T
    v8f s[4];
#pragma unroll
    for (int nt = 0; nt < 4; ++nt) {
      v8f z = v8f{};
      z = wmma_bf16(aQ0, KsF[cur][nt][0][lane], z);
      z = wmma_bf16(aQ1, KsF[cur][nt][1][lane], z);
      s[nt] = z;
    }

    // Scale + mask: one aligned 8B global load (L2-resident maskT) per tile
#pragma unroll
    for (int nt = 0; nt < 4; ++nt) {
      const unsigned long long mb = *(const unsigned long long*)(
          MgT + ((size_t)(kt * 64 + nt * 16 + ln)) * N_SEQ + wave * 16 + 8 * hi);
#pragma unroll
      for (int i = 0; i < 8; ++i) {
        const float v = s[nt][i] * scale;
        s[nt][i] = ((mb >> (8 * i)) & 0xffull) ? v : -1e30f;
      }
    }

    // Row max: tile-wise then 16-lane shuffle reduce
    float rowm[8];
#pragma unroll
    for (int i = 0; i < 8; ++i) {
      float v = fmaxf(fmaxf(s[0][i], s[1][i]), fmaxf(s[2][i], s[3][i]));
#pragma unroll
      for (int off = 8; off >= 1; off >>= 1)
        v = fmaxf(v, __shfl_xor(v, off, 32));
      rowm[i] = v;
    }

    // Online softmax update
    float alpha[8], newm[8];
#pragma unroll
    for (int i = 0; i < 8; ++i) {
      newm[i]  = fmaxf(mrow[i], rowm[i]);
      alpha[i] = __expf(mrow[i] - newm[i]);
      mrow[i]  = newm[i];
    }
    float rsum[8];
#pragma unroll
    for (int i = 0; i < 8; ++i) rsum[i] = 0.0f;
#pragma unroll
    for (int nt = 0; nt < 4; ++nt) {
#pragma unroll
      for (int i = 0; i < 8; ++i) {
        const float p = __expf(s[nt][i] - newm[i]);
        s[nt][i] = p;
        rsum[i] += p;
      }
    }
#pragma unroll
    for (int i = 0; i < 8; ++i) {
#pragma unroll
      for (int off = 8; off >= 1; off >>= 1)
        rsum[i] += __shfl_xor(rsum[i], off, 32);
      lrow[i] = lrow[i] * alpha[i] + rsum[i];
    }
#pragma unroll
    for (int t = 0; t < 4; ++t)
#pragma unroll
      for (int i = 0; i < 8; ++i) accO[t][i] *= alpha[i];

    // Write P straight into A-fragment layout (wave-private region)
#pragma unroll
    for (int nt = 0; nt < 4; ++nt) {
      const int j = nt * 16 + ln, jl = j & 31, kb = j >> 5;
      __bf16* dst = (__bf16*)&PsF[wave][kb][0];
#pragma unroll
      for (int i = 0; i < 8; ++i)
        dst[frag_off(i + 8 * hi, jl)] = (__bf16)s[nt][i];
    }

    const v16bf aP0 = PsF[wave][0][lane];
    const v16bf aP1 = PsF[wave][1][lane];

    // O += P V
#pragma unroll
    for (int t = 0; t < 4; ++t) {
      accO[t] = wmma_bf16(aP0, VsF[cur][t][0][lane], accO[t]);
      accO[t] = wmma_bf16(aP1, VsF[cur][t][1][lane], accO[t]);
    }
  }

  // Normalize and emit O as [B*N, E] (col = h*64 + d), bf16
#pragma unroll
  for (int i = 0; i < 8; ++i) {
    const float inv = 1.0f / lrow[i];
    const int row = qt * 64 + wave * 16 + i + 8 * hi;
#pragma unroll
    for (int t = 0; t < 4; ++t) {
      const int col = h * 64 + t * 16 + ln;
      Ob[((size_t)b * N_SEQ + row) * E_DIM + col] = (__bf16)(accO[t][i] * inv);
    }
  }
}

// ---------------------------------------------------------------------------
extern "C" void kernel_launch(void* const* d_in, const int* in_sizes, int n_in,
                              void* d_out, int out_size, void* d_ws, size_t ws_size,
                              hipStream_t stream) {
  const float* query = (const float*)d_in[0];
  const float* key_t = (const float*)d_in[1];
  const float* value = (const float*)d_in[2];
  const unsigned char* mask = (const unsigned char*)d_in[3];
  const float* Wq = (const float*)d_in[4];
  const float* bq = (const float*)d_in[5];
  const float* Wk = (const float*)d_in[6];
  const float* bk = (const float*)d_in[7];
  const float* Wv = (const float*)d_in[8];
  const float* bv = (const float*)d_in[9];
  const float* Wo = (const float*)d_in[10];
  const float* bo = (const float*)d_in[11];

  const int M = B_BATCH * N_SEQ;        // 8192
  const size_t tEl = (size_t)M * E_DIM; // elems per projected tensor

  char* ws = (char*)d_ws;
  __bf16* Qbf = (__bf16*)(ws);
  __bf16* Kbf = (__bf16*)(ws + 2 * tEl);
  __bf16* VbfT = (__bf16*)(ws + 4 * tEl);
  __bf16* Obf = (__bf16*)(ws + 6 * tEl);
  unsigned char* maskT = (unsigned char*)(ws + 8 * tEl);  // 16.8 MB, total ~49 MB

  dim3 gblk(128, 1, 1);
  dim3 ggrid(E_DIM / 64, M / 64, 1);      // (8, 128)

  // One-shot mask transpose (L2-resident thereafter)
  dim3 mgrid(N_SEQ / 64, N_SEQ / 64, B_BATCH);
  mask_transpose_kernel<<<mgrid, gblk, 0, stream>>>(mask, maskT);

  gemm_bias_kernel<false, 1><<<ggrid, gblk, 0, stream>>>(query, Wq, bq, Qbf);
  gemm_bias_kernel<false, 1><<<ggrid, gblk, 0, stream>>>(key_t, Wk, bk, Kbf);
  gemm_bias_kernel<false, 2><<<ggrid, gblk, 0, stream>>>(value, Wv, bv, VbfT);

  dim3 agrid(N_SEQ / 64, H_HEADS, B_BATCH); // (32, 8, 4)
  attn_kernel<<<agrid, gblk, 0, stream>>>(Qbf, Kbf, VbfT, maskT, Obf);

  gemm_bias_kernel<true, 0><<<ggrid, gblk, 0, stream>>>(Obf, Wo, bo, d_out);
}
